// CIN2_15290083573768
// MI455X (gfx1250) — matrix operand order repaired and checked by reading, not compiled, not run on hardware
//
#include <hip/hip_runtime.h>
#include <math.h>
#include <stdint.h>

// ---------------------------------------------------------------------------
// xDeepFM CIN with dynamic conv, CDNA5 (gfx1250) WMMA implementation.
//   B=512, F=39, E=128, UNITS=16x3, K=4 experts, kernel-3 conv (pad 1).
// Pipeline per layer:
//   1) ctx_attn_kernel : ctx = mean_e(Z), 2-layer MLP + softmax -> attn[b,4]
//   2) agg_kernel      : agg = sum_k attn_k * W_k, written as bf16 in the
//                        WMMA A-operand lane layout used here (k = 16h + j)
//   3) conv_wmma_kernel: per-sample GEMM [16,C]x[C,128] for 3 taps via WMMA.
//                        Fully unrolled; Z built from register-resident
//                        X0/Xk columns; LDS staged via TDM tensor_load_to_lds.
//   4) fc_kernel       : final [48]->1 projection
// ---------------------------------------------------------------------------

typedef __attribute__((ext_vector_type(16))) __bf16 v16bf;
typedef __attribute__((ext_vector_type(8)))  float  v8f;

#define NB   512
#define FDIM 39
#define EDIM 128
#define OUTU 16
#define BCH  128   // batch chunk for agg staging

// --------------------------- TDM support probes ----------------------------
#if defined(__has_builtin)
#  if __has_builtin(__builtin_amdgcn_tensor_load_to_lds)
#    define ATHENA_TDM 1
#  endif
#endif
#ifndef ATHENA_TDM
#  define ATHENA_TDM 0
#endif

#if defined(__has_include)
#  if __has_include(<hip/amd_detail/amd_gfx1250_TDM.h>)
#    define ATHENA_TDM_6ARG 1   // therock-10.0 headers -> 6-arg builtin
#  endif
#endif
#ifndef ATHENA_TDM_6ARG
#  define ATHENA_TDM_6ARG 0    // ROCm 7.2 -> 5-arg builtin
#endif

#if ATHENA_TDM
typedef unsigned int v4u __attribute__((ext_vector_type(4)));
typedef int          v4i __attribute__((ext_vector_type(4)));
typedef int          v8i __attribute__((ext_vector_type(8)));

// 1-D contiguous global->LDS DMA of nelem f32 via the Tensor Data Mover.
// D# group0: count=1 | lds_addr | global_addr(57b) | type=2.
// D# group1: data_size=4B; tensor_dim0 = tile_dim0 = nelem; tensor_dim1=1;
//            tile_dim1/2 = 0 (unused); stride0 = nelem; no pad/iterate/mask.
__device__ __forceinline__
void tdm_load_1d(unsigned lds_byte_off, const void* gsrc, unsigned nelem)
{
    unsigned long long ga = (unsigned long long)(uintptr_t)gsrc;
    v4u g0;
    g0.x = 1u;                                        // count=1, user D#
    g0.y = lds_byte_off;                              // lds_addr
    g0.z = (unsigned)(ga & 0xffffffffu);              // global_addr[31:0]
    g0.w = (unsigned)((ga >> 32) & 0x01ffffffu)       // global_addr[56:32]
         | 0x80000000u;                               // type = 2 ("image")
    v8i g1;
    g1[0] = 0x00020000;                               // data_size=2 (4 bytes)
    g1[1] = (int)((nelem & 0xffffu) << 16);           // tensor_dim0[15:0]
    g1[2] = (int)((nelem >> 16) & 0xffffu)            // tensor_dim0[31:16]
          | (1 << 16);                                // tensor_dim1 = 1
    g1[3] = (int)((nelem & 0xffffu) << 16);           // tile_dim0
    g1[4] = 0;                                        // tile_dim1/2 unused
    g1[5] = (int)nelem;                               // tensor_dim0_stride
    g1[6] = 0;
    g1[7] = 0;
    v4i z4 = {0, 0, 0, 0};
#if ATHENA_TDM_6ARG
    v8i z8 = {0, 0, 0, 0, 0, 0, 0, 0};
    __builtin_amdgcn_tensor_load_to_lds(g0, g1, z4, z4, z8, 0);
#else
    __builtin_amdgcn_tensor_load_to_lds(g0, g1, z4, z4, 0);
#endif
}

__device__ __forceinline__ void tdm_wait_all()
{
#if defined(__has_builtin) && __has_builtin(__builtin_amdgcn_s_wait_tensorcnt)
    __builtin_amdgcn_s_wait_tensorcnt(0);
#else
    asm volatile("s_wait_tensorcnt 0x0" ::: "memory");
#endif
}
#endif // ATHENA_TDM

// ---------------------------------------------------------------------------
// 1) ctx + attention MLP -> attn[b,4]
// ---------------------------------------------------------------------------
__global__ __launch_bounds__(256)
void ctx_attn_kernel(const float* __restrict__ X0g,
                     const float* __restrict__ Xkg,
                     const float* __restrict__ aw1,
                     const float* __restrict__ ab1,
                     const float* __restrict__ aw2,
                     const float* __restrict__ ab2,
                     float* __restrict__ attn_g,
                     int H, int C, int hid)
{
    __shared__ float X0s[FDIM * EDIM];
    __shared__ float Xks[FDIM * EDIM];
    __shared__ float ctx[1536];
    __shared__ float hidn[384];
    __shared__ float logit[4];

    const int b   = blockIdx.x;
    const int tid = threadIdx.x;

    for (int i = tid; i < FDIM * EDIM; i += 256) X0s[i] = X0g[(size_t)b * FDIM * EDIM + i];
    for (int i = tid; i < H * EDIM;   i += 256) Xks[i] = Xkg[(size_t)b * H * EDIM + i];
    __syncthreads();

    // ctx[c] = (1/E) * dot(X0[d,:], Xk[h,:]),  c = d*H + h
    for (int c = tid; c < C; c += 256) {
        int d  = c / H;
        int hh = c - d * H;
        const float* xa = &X0s[d * EDIM];
        const float* xb = &Xks[hh * EDIM];
        float s = 0.f;
        #pragma unroll 8
        for (int e = 0; e < EDIM; ++e) s += xa[e] * xb[e];
        ctx[c] = s * (1.0f / 128.0f);
    }
    __syncthreads();

    // hidden = relu(ctx @ aw1 + ab1)
    for (int u = tid; u < hid; u += 256) {
        float s = ab1[u];
        for (int c = 0; c < C; ++c) s += ctx[c] * aw1[(size_t)c * hid + u];
        hidn[u] = s > 0.f ? s : 0.f;
    }
    __syncthreads();

    // logits = hidden @ aw2 + ab2
    if (tid < 4) {
        float s = ab2[tid];
        for (int u = 0; u < hid; ++u) s += hidn[u] * aw2[u * 4 + tid];
        logit[tid] = s;
    }
    __syncthreads();

    if (tid == 0) {
        float m = logit[0];
        #pragma unroll
        for (int k = 1; k < 4; ++k) m = fmaxf(m, logit[k]);
        float e0 = __expf(logit[0] - m), e1 = __expf(logit[1] - m);
        float e2 = __expf(logit[2] - m), e3 = __expf(logit[3] - m);
        float inv = 1.0f / (e0 + e1 + e2 + e3);
        attn_g[b * 4 + 0] = e0 * inv;
        attn_g[b * 4 + 1] = e1 * inv;
        attn_g[b * 4 + 2] = e2 * inv;
        attn_g[b * 4 + 3] = e3 * inv;
    }
}

// ---------------------------------------------------------------------------
// 2) agg = sum_k attn_k * W[k,o,c,t]  -> bf16, pre-swizzled WMMA A layout.
//    Per (b-local, tap t, chunk q): 16x32 tile.  Lane L (m=L&15, h=L>>4),
//    element j  ->  K = 16h + j;  stored at L*16 + j.
// ---------------------------------------------------------------------------
__global__ __launch_bounds__(256)
void agg_kernel(const float* __restrict__ W,      // [4,16,C,3]
                const float* __restrict__ attn_g,
                __bf16* __restrict__ aggbuf,
                int b0, int C, int nq)
{
    __shared__ float at[4];
    const int q   = blockIdx.x;
    const int bl  = blockIdx.y;
    const int b   = b0 + bl;
    const int tid = threadIdx.x;

    if (tid < 4) at[tid] = attn_g[b * 4 + tid];
    __syncthreads();

    const size_t kstride = (size_t)16 * C * 3;     // expert stride in W
    for (int ii = tid; ii < 1536; ii += 256) {     // 3 taps * 32 lanes * 16 elems
        int t    = ii >> 9;
        int rem  = ii & 511;
        int L    = rem >> 4;
        int j    = rem & 15;
        int o    = L & 15;
        int h    = L >> 4;
        int koff = 16 * h + j;
        int c    = q * 32 + koff;
        float v = 0.f;
        if (c < C) {
            size_t base = ((size_t)o * C + c) * 3 + t;
            v = at[0] * W[base]
              + at[1] * W[base + kstride]
              + at[2] * W[base + 2 * kstride]
              + at[3] * W[base + 3 * kstride];
        }
        aggbuf[(size_t)bl * (3 * (size_t)nq * 512)
               + (size_t)((t * nq + q) * 512) + L * 16 + j] = (__bf16)v;
    }
}

// ---------------------------------------------------------------------------
// 3) Main WMMA kernel helpers: fully constant-indexed B-fragment build so the
//    per-lane X0/Xk columns stay in VGPRs (no LDS/flat traffic in the loop).
//    B layout assumption: lanes 0-15 hold K rows 0-15 (elem j -> K=j),
//    lanes 16-31 hold K rows 16-31 (elem j -> K=16+j).
// ---------------------------------------------------------------------------
template<int H, int C, bool SAME, int XKN, int Q, int J>
__device__ __forceinline__
void bfill(v16bf& bv, bool hi, const float (&x0c)[FDIM], const float (&xkc)[XKN])
{
    constexpr int clo = Q * 32 + J;        // K row for lanes 0-15
    constexpr int chi = clo + 16;          // K row for lanes 16-31
    float vlo = 0.f, vhi = 0.f;
    if constexpr (clo < C) {
        constexpr int d = clo / H, r = clo % H;
        if constexpr (SAME) vlo = x0c[d] * x0c[r];
        else                vlo = x0c[d] * xkc[r];
    }
    if constexpr (chi < C) {
        constexpr int d = chi / H, r = chi % H;
        if constexpr (SAME) vhi = x0c[d] * x0c[r];
        else                vhi = x0c[d] * xkc[r];
    }
    bv[J] = (__bf16)(hi ? vhi : vlo);
    if constexpr (J < 15) bfill<H, C, SAME, XKN, Q, J + 1>(bv, hi, x0c, xkc);
}

template<int H, int C, int NQ, bool SAME, int XKN, int Q>
__device__ __forceinline__
void chunk_run(v8f& G0, v8f& G1, v8f& G2, const v16bf* __restrict__ Ab,
               int lane, bool hi, const float (&x0c)[FDIM], const float (&xkc)[XKN])
{
    v16bf bv;
    bfill<H, C, SAME, XKN, Q, 0>(bv, hi, x0c, xkc);
    v16bf a0 = Ab[(0 * NQ + Q) * 32 + lane];
    v16bf a1 = Ab[(1 * NQ + Q) * 32 + lane];
    v16bf a2 = Ab[(2 * NQ + Q) * 32 + lane];
    G0 = __builtin_amdgcn_wmma_f32_16x16x32_bf16(false, a0, false, bv,
                                                 (short)0, G0, false, false);
    G1 = __builtin_amdgcn_wmma_f32_16x16x32_bf16(false, a1, false, bv,
                                                 (short)0, G1, false, false);
    G2 = __builtin_amdgcn_wmma_f32_16x16x32_bf16(false, a2, false, bv,
                                                 (short)0, G2, false, false);
    if constexpr (Q + 1 < NQ)
        chunk_run<H, C, NQ, SAME, XKN, Q + 1>(G0, G1, G2, Ab, lane, hi, x0c, xkc);
}

// One block per sample, 8 waves; wave w owns e-columns [16w, 16w+16).
// G_t = agg_t @ Z accumulated via WMMA, then y[o,e] = G0[o,e-1]+G1[o,e]+G2[o,e+1].
template<int H, int C, int NQ, bool SAME>
__global__ __launch_bounds__(256)
void conv_wmma_kernel(const float* __restrict__ X0g,
                      const float* __restrict__ Xkg,
                      const __bf16* __restrict__ aggbuf,
                      float* __restrict__ Xnext,     // [NB,16,128]
                      float* __restrict__ outs,      // [NB,48]
                      int b0, int layer_off)
{
    constexpr int XKN = SAME ? 1 : H;
    __shared__ float X0s[FDIM * EDIM];               // 19968 B
    __shared__ float Xks[XKN * EDIM];
    __shared__ float Gs[3][OUTU * EDIM];             // 24576 B
    __shared__ float osum[OUTU];

    const int bl  = blockIdx.x;
    const int b   = b0 + bl;
    const int tid = threadIdx.x;

    const v16bf* Ab =
        (const v16bf*)(aggbuf + (size_t)bl * (3 * (size_t)NQ * 512));

    // Prefetch this block's pre-swizzled agg stream (global_prefetch_b8).
    {
        const char* apc = (const char*)Ab;
        for (int off = tid * 128; off < 3 * NQ * 1024; off += 256 * 128)
            __builtin_prefetch(apc + off, 0, 1);
    }

#if ATHENA_TDM
    // Stage X0 (and Xk) into LDS via the Tensor Data Mover.  TDM ignores
    // EXEC; every wave that reaches the call issues its own (identical) DMA
    // and waits on its own TENSORcnt, so any lowering of this guard is safe.
    if (tid < 32) {
        tdm_load_1d((unsigned)(uintptr_t)X0s,
                    X0g + (size_t)b * FDIM * EDIM, FDIM * EDIM);
        if constexpr (!SAME)
            tdm_load_1d((unsigned)(uintptr_t)Xks,
                        Xkg + (size_t)b * H * EDIM, H * EDIM);
        tdm_wait_all();
    }
#else
    for (int i = tid; i < FDIM * EDIM; i += 256) X0s[i] = X0g[(size_t)b * FDIM * EDIM + i];
    if constexpr (!SAME)
        for (int i = tid; i < H * EDIM; i += 256) Xks[i] = Xkg[(size_t)b * H * EDIM + i];
#endif
    if (tid < OUTU) osum[tid] = 0.f;
    __syncthreads();

    const int w    = tid >> 5;          // wave id -> e-tile
    const int lane = tid & 31;
    const int h    = lane >> 4;         // K-half for B operand
    const int n    = lane & 15;         // column within e-tile
    const int e    = w * 16 + n;

    // Hoist this lane's X0 / Xk columns into registers (constant indices only).
    float x0c[FDIM];
    #pragma unroll
    for (int d = 0; d < FDIM; ++d) x0c[d] = X0s[d * EDIM + e];
    float xkc[XKN];
    if constexpr (!SAME) {
        #pragma unroll
        for (int r = 0; r < H; ++r) xkc[r] = Xks[r * EDIM + e];
    } else {
        xkc[0] = 0.f;
    }

    v8f G0 = {}, G1 = {}, G2 = {};
    chunk_run<H, C, NQ, SAME, XKN, 0>(G0, G1, G2, Ab, lane, h != 0, x0c, xkc);

    // C/D layout: VGPR r -> row M = r + 8h, column N = n
    #pragma unroll
    for (int r = 0; r < 8; ++r) {
        int m = r + 8 * h;
        Gs[0][m * EDIM + e] = G0[r];
        Gs[1][m * EDIM + e] = G1[r];
        Gs[2][m * EDIM + e] = G2[r];
    }
    __syncthreads();

    // tap shift-combine + store + sum over e
    for (int idx = tid; idx < OUTU * EDIM; idx += 256) {
        int o  = idx >> 7;
        int ee = idx & 127;
        float y = Gs[1][o * EDIM + ee];
        if (ee > 0)   y += Gs[0][o * EDIM + ee - 1];
        if (ee < 127) y += Gs[2][o * EDIM + ee + 1];
        Xnext[((size_t)b * OUTU + o) * EDIM + ee] = y;
        atomicAdd(&osum[o], y);                    // ds_add_f32
    }
    __syncthreads();
    if (tid < OUTU) outs[(size_t)b * 48 + layer_off + tid] = osum[tid];
}

// ---------------------------------------------------------------------------
// 4) final FC: out[b] = outs[b,:] @ fc_w + fc_b
// ---------------------------------------------------------------------------
__global__ __launch_bounds__(256)
void fc_kernel(const float* __restrict__ outs,
               const float* __restrict__ fc_w,
               const float* __restrict__ fc_b,
               float* __restrict__ out)
{
    int b = blockIdx.x * 256 + threadIdx.x;
    if (b >= NB) return;
    float s = fc_b[0];
    #pragma unroll
    for (int j = 0; j < 48; ++j) s += outs[b * 48 + j] * fc_w[j];
    out[b] = s;
}

// ---------------------------------------------------------------------------
extern "C" void kernel_launch(void* const* d_in, const int* in_sizes, int n_in,
                              void* d_out, int out_size, void* d_ws, size_t ws_size,
                              hipStream_t stream)
{
    (void)in_sizes; (void)n_in; (void)out_size; (void)ws_size;

    const float* X0g = (const float*)d_in[0];

    // workspace layout (all offsets 256B aligned)
    char*  ws    = (char*)d_ws;
    float* attn  = (float*)(ws + 0);            //   8192 B : [512,4]
    float* outs  = (float*)(ws + 8192);         //  98304 B : [512,48]
    float* XkA   = (float*)(ws + 106496);       //    4 MB : [512,16,128]
    float* XkB   = (float*)(ws + 4300800);      //    4 MB
    __bf16* aggb = (__bf16*)(ws + 8495104);     // <=18.9 MB staged agg (bf16)

    const float* xin[3]  = { X0g, XkA, XkB };
    const int Hs[3]   = { 39, 16, 16 };
    const int Cs[3]   = { 39 * 39, 39 * 16, 39 * 16 };   // 1521, 624, 624
    const int hids[3] = { 380, 156, 156 };               // int(C*0.25)
    const int nqs[3]  = { 48, 20, 20 };                  // ceil(C/32)

    for (int l = 0; l < 3; ++l) {
        const float* W   = (const float*)d_in[1 + l * 5 + 0];
        const float* aw1 = (const float*)d_in[1 + l * 5 + 1];
        const float* ab1 = (const float*)d_in[1 + l * 5 + 2];
        const float* aw2 = (const float*)d_in[1 + l * 5 + 3];
        const float* ab2 = (const float*)d_in[1 + l * 5 + 4];

        ctx_attn_kernel<<<NB, 256, 0, stream>>>(X0g, xin[l], aw1, ab1, aw2, ab2,
                                                attn, Hs[l], Cs[l], hids[l]);

        for (int b0 = 0; b0 < NB; b0 += BCH) {
            dim3 ga(nqs[l], BCH);
            agg_kernel<<<ga, 256, 0, stream>>>(W, attn, aggb, b0, Cs[l], nqs[l]);
            if (l == 0) {
                conv_wmma_kernel<39, 1521, 48, true>
                    <<<BCH, 256, 0, stream>>>(X0g, X0g, aggb, XkA, outs, b0, 0);
            } else if (l == 1) {
                conv_wmma_kernel<16, 624, 20, false>
                    <<<BCH, 256, 0, stream>>>(X0g, XkA, aggb, XkB, outs, b0, 16);
            } else {
                conv_wmma_kernel<16, 624, 20, false>
                    <<<BCH, 256, 0, stream>>>(X0g, XkB, aggb, XkA, outs, b0, 32);
            }
        }
    }

    fc_kernel<<<2, 256, 0, stream>>>(outs, (const float*)d_in[16],
                                     (const float*)d_in[17], (float*)d_out);
}
